// JointAttention_73315091743104
// MI455X (gfx1250) — matrix operand order repaired and checked by reading, not compiled, not run on hardware
//
#include <hip/hip_runtime.h>
#include <hip/hip_bf16.h>
#include <math.h>

typedef __attribute__((ext_vector_type(16))) _Float16 v16h;
typedef __attribute__((ext_vector_type(8)))  float    v8f;

#define HH  16
#define DHD 64

// ---------------------------------------------------------------------------
// Async global->LDS copy (CDNA5 TDM-adjacent path, tracked by ASYNCcnt).
// lds_off is the low-32-bits of the generic pointer (== LDS byte offset).
// ---------------------------------------------------------------------------
__device__ __forceinline__ void async_copy_b128(unsigned lds_off, const void* gptr)
{
    asm volatile("global_load_async_to_lds_b128 %0, %1, off"
                 :: "v"(lds_off), "v"((unsigned long long)(size_t)gptr)
                 : "memory");
}
__device__ __forceinline__ void wait_async0()
{
    asm volatile("s_wait_asynccnt 0x0" ::: "memory");
}

// ---------------------------------------------------------------------------
// Generic GEMM: C[M,N] = A[M,K] * W[K,N] (+ bias), fp32 in/out, f16 WMMA core.
// Block = 256 threads = 8 waves; block tile 128x64, each wave owns a 32x32
// sub-tile -> 4 WMMAs per K32 step from 2 A-frags + 2 B-frags (reuse).
// A tile staged via global_load_async_to_lds_b128 (fp32 in LDS, cvt on read);
// W tile staged via VGPRs with on-the-fly transpose to f16 [n][k].
// ---------------------------------------------------------------------------
#define BM 128
#define BN 64
#define BK 32

__global__ __launch_bounds__(256)
void gemm_wmma_f16(const float* __restrict__ A, const float* __restrict__ W,
                   const float* __restrict__ bias, float* __restrict__ C,
                   int M, int N, int K)
{
    __shared__ float    Asf[BM][BK + 4];   // fp32 [m][k], row=36 floats (bank-safe)
    __shared__ _Float16 Bs [BN][BK + 4];   // f16 [n][k]  (W tile stored transposed)

    const int tid  = threadIdx.x;
    const int lane = tid & 31;
    const int w    = tid >> 5;         // 0..7
    const int wm   = (w >> 1) * 32;    // 0,32,64,96
    const int wn   = (w & 1) * 32;     // 0,32
    const int gm0  = blockIdx.y * BM;
    const int gn0  = blockIdx.x * BN;

    // A async staging: 2 threads per row, 4x 16B chunks each
    const int arow = tid >> 1;
    const int acol = (tid & 1) * 16;           // float index
    // B staging: 32 rows x 64 cols; thread: k-row tid>>3, 8 cols at (tid&7)*8
    const int kr = tid >> 3;
    const int nc = (tid & 7) * 8;

    v8f acc00 = {}, acc01 = {}, acc10 = {}, acc11 = {};

    for (int kb = 0; kb < K; kb += BK) {
        __syncthreads();
        // --- stage A tile: async global->LDS, fp32, no VGPR round trip ---
        const float*   gsrc = &A[(size_t)(gm0 + arow) * K + kb + acol];
        const unsigned ldsa = (unsigned)(size_t)&Asf[arow][acol];
#pragma unroll
        for (int c = 0; c < 4; ++c)
            async_copy_b128(ldsa + 16u * c, gsrc + 4 * c);

        // --- stage W tile transposed: Bs[n][k] (f16) ---
        const float4* pw =
            reinterpret_cast<const float4*>(&W[(size_t)(kb + kr) * N + gn0 + nc]);
        float4 u0 = pw[0], u1 = pw[1];
        float vw[8] = {u0.x, u0.y, u0.z, u0.w, u1.x, u1.y, u1.z, u1.w};
#pragma unroll
        for (int j = 0; j < 8; ++j) Bs[nc + j][kr] = (_Float16)vw[j];

        // prefetch next K tile into cache while this one is consumed
        if (kb + BK < K) {
            __builtin_prefetch(&A[(size_t)(gm0 + arow) * K + kb + BK + acol], 0, 0);
            __builtin_prefetch(&W[(size_t)(kb + BK + kr) * N + gn0 + nc], 0, 0);
        }

        wait_async0();          // this wave's async tile chunks are in LDS
        __syncthreads();        // everyone's chunks visible

        // --- fragments in WMMA wave32 register layout ---
        const int ml = lane & 15;
        const int ab = (lane < 16) ? 0 : 8;    // A: K base 0 / 8
        const int bb = (lane < 16) ? 0 : 16;   // B: K base 0 / 16
        v16h a0, a1, b0, b1;
#pragma unroll
        for (int i = 0; i < 16; ++i) {
            const int ka = 2 * ab + i;   // K = 16*half + i (contiguous)
            a0[i] = (_Float16)Asf[wm + ml][ka];
            a1[i] = (_Float16)Asf[wm + 16 + ml][ka];
            b0[i] = Bs[wn + ml][bb + i];                    // K contiguous per half
            b1[i] = Bs[wn + 16 + ml][bb + i];
        }
        acc00 = __builtin_amdgcn_wmma_f32_16x16x32_f16(false, a0, false, b0,
                                                       (short)0, acc00, false, false);
        acc01 = __builtin_amdgcn_wmma_f32_16x16x32_f16(false, a0, false, b1,
                                                       (short)0, acc01, false, false);
        acc10 = __builtin_amdgcn_wmma_f32_16x16x32_f16(false, a1, false, b0,
                                                       (short)0, acc10, false, false);
        acc11 = __builtin_amdgcn_wmma_f32_16x16x32_f16(false, a1, false, b1,
                                                       (short)0, acc11, false, false);
    }

    // --- epilogue: C-fragment layout -> global (row = r + 8*laneHi) ---
    const int ml = lane & 15;
    const int rb = (lane >> 4) << 3;
#pragma unroll
    for (int ti = 0; ti < 2; ++ti) {
#pragma unroll
        for (int tj = 0; tj < 2; ++tj) {
            const v8f acc = (ti == 0) ? ((tj == 0) ? acc00 : acc01)
                                      : ((tj == 0) ? acc10 : acc11);
            const int col   = gn0 + wn + tj * 16 + ml;
            const int rbase = gm0 + wm + ti * 16 + rb;
            const float bv  = bias ? bias[col] : 0.0f;
#pragma unroll
            for (int r = 0; r < 8; ++r)
                C[(size_t)(rbase + r) * N + col] = acc[r] + bv;
        }
    }
}

// ---------------------------------------------------------------------------
// Per-head RMS-norm (+ optional RoPE) and pack to f16 [B,H,dst_seq,DH].
// One wave per head; lane handles the even/odd pair (matches rotate_half).
// ---------------------------------------------------------------------------
__global__ __launch_bounds__(512)
void qk_norm_rope(const float* __restrict__ proj, const float* __restrict__ normw,
                  const float* __restrict__ freqs, _Float16* __restrict__ dst,
                  int nsrc, int dst_seq, int n_off, int use_rope)
{
    const int row = blockIdx.x;              // b*nsrc + n
    const int b   = row / nsrc;
    const int n   = row % nsrc;
    const int h   = threadIdx.x >> 5;
    const int l   = threadIdx.x & 31;
    const int d0  = 2 * l;

    const size_t src = (size_t)row * (HH * DHD) + h * DHD + d0;
    float x0 = proj[src];
    float x1 = proj[src + 1];

    float ss = x0 * x0 + x1 * x1;
    ss += __shfl_xor(ss, 16, 32);
    ss += __shfl_xor(ss, 8, 32);
    ss += __shfl_xor(ss, 4, 32);
    ss += __shfl_xor(ss, 2, 32);
    ss += __shfl_xor(ss, 1, 32);
    const float rms = rsqrtf(ss * (1.0f / DHD) + 1e-6f);

    float y0 = x0 * rms * normw[h * DHD + d0];
    float y1 = x1 * rms * normw[h * DHD + d0 + 1];

    if (use_rope) {
        const float f = freqs[(size_t)row * DHD + d0];  // repeated pairwise
        const float c = __cosf(f), s = __sinf(f);
        const float z0 = y0 * c - y1 * s;
        const float z1 = y1 * c + y0 * s;
        y0 = z0; y1 = z1;
    }

    const size_t o = (((size_t)b * HH + h) * dst_seq + (n_off + n)) * DHD + d0;
    dst[o]     = (_Float16)y0;
    dst[o + 1] = (_Float16)y1;
}

// Pack V projection fp32 [B*nsrc, H*DH] -> f16 TRANSPOSED [B,H,DH,NK] at n_off,
// so flash-attention B-fragments for P*V are contiguous 32B vector loads.
__global__ void v_pack(const float* __restrict__ proj, _Float16* __restrict__ dst,
                       int nsrc, int dst_seq, int n_off, int total)
{
    for (int i = blockIdx.x * blockDim.x + threadIdx.x; i < total;
         i += gridDim.x * blockDim.x) {
        const int d  = i & (DHD - 1);
        const int hh = (i >> 6) & (HH - 1);
        const int rn = i >> 10;
        const int b  = rn / nsrc;
        const int n  = rn % nsrc;
        dst[(((size_t)b * HH + hh) * DHD + d) * dst_seq + n_off + n] =
            (_Float16)proj[i];
    }
}

// ---------------------------------------------------------------------------
// Flash attention: one wave per (b, h, 16-query tile). Online softmax; WMMA
// for both Q*K^T and P*V. P is re-laid-out C-frag -> A-frag through LDS.
// K is [B,H,NK,DH]; V is TRANSPOSED [B,H,DH,NK] so both fragment types load
// as aligned 32-byte v16h vectors (2x global_load_b128 each).
// ---------------------------------------------------------------------------
__global__ __launch_bounds__(32)
void flash_attn(const _Float16* __restrict__ qh, const _Float16* __restrict__ kh,
                const _Float16* __restrict__ vt, const unsigned char* __restrict__ amask,
                float* __restrict__ out, int NQ, int NK)
{
    __shared__ _Float16 Plds[16][36];

    const int lane = threadIdx.x;
    const int qt = blockIdx.x, h = blockIdx.y, b = blockIdx.z;
    const int m  = lane & 15;
    const int hi = lane >> 4;
    const int ab = hi ? 8 : 0;    // A-frag K base
    const int bb = hi ? 16 : 0;   // B-frag K base

    // Q tile as two A-fragments (DH = 64 = 2 x K32)
    const size_t qbase = (((size_t)b * HH + h) * NQ + qt * 16 + m) * DHD;
    v16h qf[2];
#pragma unroll
    for (int d = 0; d < 2; ++d)
#pragma unroll
        for (int i = 0; i < 16; ++i) {
            const int dh = d * 32 + 2 * ab + i;
            qf[d][i] = qh[qbase + dh];
        }

    v8f O0 = {}, O1 = {}, O2 = {}, O3 = {};
    float mrow[8], lrow[8];
#pragma unroll
    for (int r = 0; r < 8; ++r) { mrow[r] = -1e30f; lrow[r] = 0.0f; }

    const size_t khb = ((size_t)b * HH + h) * NK;       // K rows base (keys)
    const size_t vtb = ((size_t)b * HH + h) * DHD;      // V^T rows base (dh)

    for (int k0 = 0; k0 < NK; k0 += 32) {
        v8f S[2];
#pragma unroll
        for (int sh = 0; sh < 2; ++sh) {
            const int key = k0 + sh * 16 + m;     // this lane's key column
            v8f s = {};
#pragma unroll
            for (int d = 0; d < 2; ++d) {
                // 16 contiguous halves, 32B aligned -> 2x global_load_b128
                const v16h kf =
                    *reinterpret_cast<const v16h*>(&kh[(khb + key) * DHD + d * 32 + bb]);
                s = __builtin_amdgcn_wmma_f32_16x16x32_f16(false, qf[d], false, kf,
                                                           (short)0, s, false, false);
            }
#pragma unroll
            for (int r = 0; r < 8; ++r) {
                const int qrow = qt * 16 + r + hi * 8;
                const float sv = s[r] * 0.125f;   // 1/sqrt(64)
                const unsigned char mk = amask[((size_t)b * NQ + qrow) * NK + key];
                s[r] = mk ? sv : -1e9f;
            }
            S[sh] = s;
        }

        // online softmax update (row reductions across each 16-lane half)
        float alpha[8];
#pragma unroll
        for (int r = 0; r < 8; ++r) {
            float tv = fmaxf(S[0][r], S[1][r]);
            tv = fmaxf(tv, __shfl_xor(tv, 1, 16));
            tv = fmaxf(tv, __shfl_xor(tv, 2, 16));
            tv = fmaxf(tv, __shfl_xor(tv, 4, 16));
            tv = fmaxf(tv, __shfl_xor(tv, 8, 16));
            const float mn = fmaxf(mrow[r], tv);
            alpha[r] = __expf(mrow[r] - mn);
            mrow[r]  = mn;
            const float p0 = __expf(S[0][r] - mn);
            const float p1 = __expf(S[1][r] - mn);
            float sum = p0 + p1;
            sum += __shfl_xor(sum, 1, 16);
            sum += __shfl_xor(sum, 2, 16);
            sum += __shfl_xor(sum, 4, 16);
            sum += __shfl_xor(sum, 8, 16);
            lrow[r] = lrow[r] * alpha[r] + sum;
            O0[r] *= alpha[r]; O1[r] *= alpha[r];
            O2[r] *= alpha[r]; O3[r] *= alpha[r];
            // stage P (C-frag layout) into LDS as a 16x32 row-major tile
            Plds[r + hi * 8][m]      = (_Float16)p0;
            Plds[r + hi * 8][16 + m] = (_Float16)p1;
        }
        __syncthreads();

        // read P back as an A-fragment (16x32 over the 32 keys)
        v16h pf;
#pragma unroll
        for (int i = 0; i < 16; ++i) {
            const int kk = 2 * ab + i;
            pf[i] = Plds[m][kk];
        }

        // O += P * V   (4 B-fragments cover DH = 64; V^T rows are contiguous)
#pragma unroll
        for (int j = 0; j < 4; ++j) {
            const v16h vf =
                *reinterpret_cast<const v16h*>(&vt[(vtb + j * 16 + m) * NK + k0 + bb]);
            v8f* Op = (j == 0) ? &O0 : (j == 1) ? &O1 : (j == 2) ? &O2 : &O3;
            *Op = __builtin_amdgcn_wmma_f32_16x16x32_f16(false, pf, false, vf,
                                                         (short)0, *Op, false, false);
        }
        __syncthreads();
    }

    // normalize and write to [b, q, h*64 + dh]
#pragma unroll
    for (int r = 0; r < 8; ++r) {
        const int qrow = qt * 16 + r + hi * 8;
        const float inv = 1.0f / lrow[r];
        const size_t ob = ((size_t)b * NQ + qrow) * (HH * DHD) + h * DHD;
        out[ob + 0 * 16 + m] = O0[r] * inv;
        out[ob + 1 * 16 + m] = O1[r] * inv;
        out[ob + 2 * 16 + m] = O2[r] * inv;
        out[ob + 3 * 16 + m] = O3[r] * inv;
    }
}

// out = attn * sigmoid(gate), in place
__global__ void gate_mul(float* __restrict__ attnb, const float* __restrict__ gatep,
                         int total)
{
    for (int i = blockIdx.x * blockDim.x + threadIdx.x; i < total;
         i += gridDim.x * blockDim.x) {
        const float g = gatep[i];
        attnb[i] = attnb[i] * (1.0f / (1.0f + __expf(-g)));
    }
}

__global__ void apply_mask(float* __restrict__ out, const unsigned char* __restrict__ mask,
                           int D, int total)
{
    for (int i = blockIdx.x * blockDim.x + threadIdx.x; i < total;
         i += gridDim.x * blockDim.x) {
        if (!mask[i / D]) out[i] = 0.0f;
    }
}

// ---------------------------------------------------------------------------
extern "C" void kernel_launch(void* const* d_in, const int* in_sizes, int n_in,
                              void* d_out, int out_size, void* d_ws, size_t ws_size,
                              hipStream_t stream)
{
    const int B = 2, NQ = 1024, NREF = 512, NTEXT = 256, D = 1024;
    const int NK = NQ + NREF + NTEXT;   // 1792

    const float*         x      = (const float*)d_in[0];
    const float*         refs   = (const float*)d_in[1];
    const float*         text   = (const float*)d_in[2];
    const unsigned char* mask   = (const unsigned char*)d_in[3];
    const unsigned char* amask  = (const unsigned char*)d_in[4];
    const float*         freqs  = (const float*)d_in[5];
    const float* Wq      = (const float*)d_in[6];
    const float* Wk_self = (const float*)d_in[7];
    const float* Wv_self = (const float*)d_in[8];
    const float* Wk_ref  = (const float*)d_in[9];
    const float* Wv_ref  = (const float*)d_in[10];
    const float* Wk_text = (const float*)d_in[11];
    const float* Wv_text = (const float*)d_in[12];
    const float* Wgate   = (const float*)d_in[13];
    const float* Wout    = (const float*)d_in[14];
    const float* bq      = (const float*)d_in[15];
    const float* bk_self = (const float*)d_in[16];
    const float* bv_self = (const float*)d_in[17];
    const float* bk_ref  = (const float*)d_in[18];
    const float* bv_ref  = (const float*)d_in[19];
    const float* bk_text = (const float*)d_in[20];
    const float* bv_text = (const float*)d_in[21];
    const float* q_norm_w       = (const float*)d_in[22];
    const float* k_norm_w       = (const float*)d_in[23];
    const float* k_norm_cross_w = (const float*)d_in[24];

    char* ws = (char*)d_ws;
    float*    proj  = (float*)(ws);                          //  8 MB fp32 staging
    float*    gatep = (float*)(ws + ((size_t)8  << 20));     //  8 MB
    float*    attnb = (float*)(ws + ((size_t)16 << 20));     //  8 MB
    _Float16* qhb   = (_Float16*)(ws + ((size_t)24 << 20));  //  4 MB f16 [B,H,NQ,DH]
    _Float16* khb   = (_Float16*)(ws + ((size_t)28 << 20));  //  7 MB f16 [B,H,NK,DH]
    _Float16* vtb   = (_Float16*)(ws + ((size_t)35 << 20));  //  7 MB f16 [B,H,DH,NK]

    auto gemm = [&](const float* A, const float* W, const float* bias, float* C,
                    int M, int N, int K) {
        dim3 g(N / BN, M / BM);
        gemm_wmma_f16<<<g, 256, 0, stream>>>(A, W, bias, C, M, N, K);
    };

    // Q
    gemm(x, Wq, bq, proj, B * NQ, D, D);
    qk_norm_rope<<<B * NQ, 512, 0, stream>>>(proj, q_norm_w, freqs, qhb, NQ, NQ, 0, 1);
    // K/V self
    gemm(x, Wk_self, bk_self, proj, B * NQ, D, D);
    qk_norm_rope<<<B * NQ, 512, 0, stream>>>(proj, k_norm_w, freqs, khb, NQ, NK, 0, 1);
    gemm(x, Wv_self, bv_self, proj, B * NQ, D, D);
    v_pack<<<2048, 256, 0, stream>>>(proj, vtb, NQ, NK, 0, B * NQ * D);
    // K/V ref
    gemm(refs, Wk_ref, bk_ref, proj, B * NREF, D, D);
    qk_norm_rope<<<B * NREF, 512, 0, stream>>>(proj, k_norm_cross_w, nullptr, khb,
                                               NREF, NK, NQ, 0);
    gemm(refs, Wv_ref, bv_ref, proj, B * NREF, D, D);
    v_pack<<<1024, 256, 0, stream>>>(proj, vtb, NREF, NK, NQ, B * NREF * D);
    // K/V text
    gemm(text, Wk_text, bk_text, proj, B * NTEXT, D, D);
    qk_norm_rope<<<B * NTEXT, 512, 0, stream>>>(proj, k_norm_cross_w, nullptr, khb,
                                                NTEXT, NK, NQ + NREF, 0);
    gemm(text, Wv_text, bv_text, proj, B * NTEXT, D, D);
    v_pack<<<512, 256, 0, stream>>>(proj, vtb, NTEXT, NK, NQ + NREF, B * NTEXT * D);
    // gate projection
    gemm(x, Wgate, nullptr, gatep, B * NQ, D, D);
    // attention
    flash_attn<<<dim3(NQ / 16, HH, B), 32, 0, stream>>>(qhb, khb, vtb, amask, attnb,
                                                        NQ, NK);
    // gate + output projection + mask
    gate_mul<<<2048, 256, 0, stream>>>(attnb, gatep, B * NQ * D);
    gemm(attnb, Wout, nullptr, (float*)d_out, B * NQ, D, D);
    apply_mask<<<2048, 256, 0, stream>>>((float*)d_out, mask, D, B * NQ * D);
}
